// CRF_25065429139541
// MI455X (gfx1250) — matrix operand order repaired and checked by reading, not compiled, not run on hardware
//
#include <hip/hip_runtime.h>
#include <hip/hip_bf16.h>
#include <math.h>

#define TAGSET 128
#define SSTART 126
#define SEND   127
#define NEGV   -10000.0f
#define BATCH  256
#define SEQ    512
#define BT     16   // batch tile per block (== WMMA N)

typedef __attribute__((ext_vector_type(16))) _Float16 v16h;
typedef __attribute__((ext_vector_type(8)))  _Float16 v8h;
typedef __attribute__((ext_vector_type(8)))  float    v8f;

// ---------------------------------------------------------------------------
// Kernel 1: gold path score per batch (light gather work).
// ---------------------------------------------------------------------------
__global__ void crf_gold_kernel(const float* __restrict__ feats,
                                const long long* __restrict__ tags,
                                const int* __restrict__ mask,
                                const float* __restrict__ trans,
                                float* __restrict__ gold) {
  int b = blockIdx.x * blockDim.x + threadIdx.x;
  if (b >= BATCH) return;
  int prev = SSTART;
  float score = 0.0f;
  int len = 0;
  for (int s = 0; s < SEQ; ++s) {
    int   cur = (int)tags[(size_t)b * SEQ + s];
    int   m   = mask[(size_t)b * SEQ + s];
    float mf  = (float)m;
    float tv  = trans[prev * TAGSET + cur];
    float ev  = feats[((size_t)b * SEQ + s) * TAGSET + cur];
    score += (tv + ev) * mf;
    len  += m;
    prev  = cur;
  }
  int last = (len > 0) ? (int)tags[(size_t)b * SEQ + (len - 1)] : SSTART;
  score += trans[last * TAGSET + SEND];
  gold[b] = score;
}

// ---------------------------------------------------------------------------
// Kernel 2: forward log-partition via WMMA in exp-space, then out = logZ-gold.
// One block per 16-batch tile; 8 waves, wave w owns output rows j=16w..16w+15.
// ---------------------------------------------------------------------------
__launch_bounds__(256)
__global__ void crf_fwd_kernel(const float* __restrict__ feats,
                               const float* __restrict__ trans,
                               const float* __restrict__ gold,
                               float* __restrict__ out) {
  __shared__ __attribute__((aligned(32))) float    alpha_lds[BT * TAGSET];
  __shared__ __attribute__((aligned(32))) _Float16 p_lds[BT * TAGSET];
  __shared__ float m_lds[BT];

  const int tid  = threadIdx.x;
  const int wave = tid >> 5;
  const int lane = tid & 31;
  const int hi   = lane >> 4;     // 0 | 1 (half-wave)
  const int ln   = lane & 15;
  const int b0   = blockIdx.x * BT;

  // ---- Preload A fragments: E[j][k] = exp(trans[j][k]) as f16, kept in VGPRs.
  // A 16x32 f16 layout: lane -> M=ln (+hi selects K run offset by 16);
  // halves 0..7 -> K = 32c + 16*hi + h, halves 8..15 -> K = 32c + 16*hi + 8 + h.
  v16h a_frag[4];
  {
    const int j = wave * 16 + ln;
    const float* trow = trans + j * TAGSET;
    for (int c = 0; c < 4; ++c) {
      v16h a;
      const int kb1 = 32 * c + hi * 16;
      const int kb2 = 32 * c + hi * 16 + 8;
      for (int h = 0; h < 8; ++h) {
        a[h]     = (_Float16)__expf(trow[kb1 + h]);
        a[h + 8] = (_Float16)__expf(trow[kb2 + h]);
      }
      a_frag[c] = a;
    }
  }

  // ---- Init alpha: NEG everywhere except START tag.
  {
    const int b = tid >> 4;
    const int g = tid & 15;
    for (int i = 0; i < 8; ++i) {
      const int k = g * 8 + i;
      alpha_lds[b * TAGSET + k] = (k == SSTART) ? 0.0f : NEGV;
    }
  }
  __syncthreads();

  // WMMA-phase lane mapping: batch column nb, 8 consecutive output rows from jb.
  const int nb = ln;
  const int jb = wave * 16 + hi * 8;
  const float* fptr = feats + ((size_t)(b0 + nb) * SEQ) * TAGSET + jb;

  v8f fcur = *(const v8f*)fptr;   // feats for t = 0 (8 consecutive tags)

  for (int t = 0; t < SEQ; ++t) {
    // ---- Phase 1: per-batch max over alpha, then p = exp(alpha - m) as f16.
    {
      const int b = tid >> 4;
      const int g = tid & 15;
      const float* arow = &alpha_lds[b * TAGSET + g * 8];
      float av[8];
      float mx = -3.0e38f;
      for (int i = 0; i < 8; ++i) { av[i] = arow[i]; mx = fmaxf(mx, av[i]); }
      for (int off = 1; off < 16; off <<= 1)
        mx = fmaxf(mx, __shfl_xor(mx, off, 32));
      if (g == 0) m_lds[b] = mx;
      v8h ph;
      for (int i = 0; i < 8; ++i) ph[i] = (_Float16)__expf(av[i] - mx);
      *(v8h*)&p_lds[b * TAGSET + g * 8] = ph;
    }
    __syncthreads();

    // ---- Prefetch next step's feats (register + L2 prefetch one more ahead).
    v8f fnext = fcur;
    if (t + 1 < SEQ) {
      fnext = *(const v8f*)(fptr + (size_t)(t + 1) * TAGSET);
      if (t + 2 < SEQ)
        __builtin_prefetch(fptr + (size_t)(t + 2) * TAGSET, 0, 0);
    }

    // ---- Phase 2: Y(128x16) = E @ P^T via v_wmma_f32_16x16x32_f16.
    // B 32x16 f16 layout: lane -> N=ln, 16 consecutive K from 32c + 16*hi.
    v8f acc = {};
    for (int c = 0; c < 4; ++c) {
      v16h bfrag = *(const v16h*)&p_lds[nb * TAGSET + 32 * c + hi * 16];
      acc = __builtin_amdgcn_wmma_f32_16x16x32_f16(
          false, a_frag[c], false, bfrag, (short)0, acc, false, false);
    }

    // ---- Phase 3: alpha'[nb][jb+r] = feat + m + log(Y); write back to LDS.
    const float mb = m_lds[nb];
    float* adst = &alpha_lds[nb * TAGSET + jb];
    for (int r = 0; r < 8; ++r) {
      const float y = fmaxf(acc[r], 1e-30f);  // dead rows -> ~m-69, negligible
      adst[r] = fcur[r] + mb + __logf(y);
    }
    __syncthreads();
    fcur = fnext;
  }

  // ---- Finalize: logZ = logsumexp_j(alpha[b][j] + trans[j][END]); write out.
  {
    const int b = tid >> 4;
    const int g = tid & 15;
    float v[8];
    float mx = -3.0e38f;
    for (int i = 0; i < 8; ++i) {
      const int k = g * 8 + i;
      v[i] = alpha_lds[b * TAGSET + k] + trans[k * TAGSET + SEND];
      mx = fmaxf(mx, v[i]);
    }
    for (int off = 1; off < 16; off <<= 1)
      mx = fmaxf(mx, __shfl_xor(mx, off, 32));
    float s = 0.0f;
    for (int i = 0; i < 8; ++i) s += __expf(v[i] - mx);
    for (int off = 1; off < 16; off <<= 1)
      s += __shfl_xor(s, off, 32);
    if (g == 0)
      out[b0 + b] = (mx + __logf(s)) - gold[b0 + b];
  }
}

// ---------------------------------------------------------------------------
extern "C" void kernel_launch(void* const* d_in, const int* in_sizes, int n_in,
                              void* d_out, int out_size, void* d_ws, size_t ws_size,
                              hipStream_t stream) {
  (void)in_sizes; (void)n_in; (void)out_size; (void)ws_size;
  const float*     feats = (const float*)d_in[0];
  const long long* tags  = (const long long*)d_in[1];
  const int*       mask  = (const int*)d_in[2];
  const float*     trans = (const float*)d_in[3];
  float* out  = (float*)d_out;
  float* gold = (float*)d_ws;   // 256 floats of scratch

  crf_gold_kernel<<<1, 256, 0, stream>>>(feats, tags, mask, trans, gold);
  crf_fwd_kernel<<<BATCH / BT, 256, 0, stream>>>(feats, trans, gold, out);
}
